// CondNetCSLoss_57277683860141
// MI455X (gfx1250) — compile-verified, matrix operand order checked by reading, not run on hardware
//
#include <hip/hip_runtime.h>
#include <hip/hip_bf16.h>
#include <stdint.h>

// Problem constants (from reference setup_inputs)
#define B_   16
#define H_   512
#define W_   512
#define HW_  262144            // H_*W_
#define NREG 48                // B_*3 regions
#define WPR  16                // u32 words per row (512 bits)
#define WORDS_REGION 8192      // 512 rows * 16 words

typedef __attribute__((ext_vector_type(2)))  float    v2f;
typedef __attribute__((ext_vector_type(8)))  float    v8f;
typedef __attribute__((ext_vector_type(16))) _Float16 v16h;

// ---------------------------------------------------------------------------
// Wave-level f32 sum using WMMA against an all-ones B matrix.
//  A(16x4 f32): lane m supplies A[m][0]=v (vgpr0), A[m][1]=0; lane m+16
//  supplies A[m][2]=v, A[m][3]=0.  With B==1 everywhere:
//    D[m][n] = v(lane m) + v(lane m+16)   (exact, f32)
//  Lane L<16 holds D[v][L] in vgpr v (M=0..7); lanes 16..31 hold M=8..15.
//  Sum of the 8 D vgprs per lane + shfl_xor(16) yields the full 32-lane sum.
// ---------------------------------------------------------------------------
__device__ __forceinline__ float wave_reduce_sum(float v) {
#if __has_builtin(__builtin_amdgcn_wmma_f32_16x16x4_f32)
  v2f a; a[0] = v; a[1] = 0.0f;
  v2f bO; bO[0] = 1.0f; bO[1] = 1.0f;
  v8f c = {};
  v8f d = __builtin_amdgcn_wmma_f32_16x16x4_f32(false, a, false, bO,
                                                (short)0, c, false, false);
#else
  // Fallback: codegen-confirmed f16 WMMA.  Row m gets lane m's element at K=0
  // and lane m+16's element at K=8; with B==1: D[m][n] = v(m)+v(m+16).
  v16h a = {};
  a[0] = (_Float16)v;
  v16h bO;
  #pragma unroll
  for (int i = 0; i < 16; ++i) bO[i] = (_Float16)1.0f;
  v8f c = {};
  v8f d = __builtin_amdgcn_wmma_f32_16x16x32_f16(false, a, false, bO,
                                                 (short)0, c, false, false);
#endif
  float s = d[0] + d[1] + d[2] + d[3] + d[4] + d[5] + d[6] + d[7];
  s += __shfl_xor(s, 16, 32);
  return s;
}

// Deterministic block reduction (valid in thread 0).  Must be called by all
// threads of a full block (WMMA requires EXEC == all ones).
__device__ __forceinline__ float block_reduce_sum(float v, float* red) {
  float w = wave_reduce_sum(v);
  int lane = threadIdx.x & 31;
  int wid  = threadIdx.x >> 5;
  int nw   = (int)(blockDim.x >> 5);
  if (lane == 0) red[wid] = w;
  __syncthreads();
  float s = 0.0f;
  if (threadIdx.x == 0) {
    for (int i = 0; i < nw; ++i) s += red[i];
  }
  __syncthreads();
  return s;
}

// ---------------------------------------------------------------------------
// Async global->LDS staging of a 32KB region bitmap (CDNA5 async path).
// ---------------------------------------------------------------------------
__device__ __forceinline__ void async_copy_bitmap(unsigned* lds, const unsigned* src) {
  uint32_t ldsBase = (uint32_t)(uintptr_t)lds;  // low 32 bits == LDS byte offset
  for (int i = (int)threadIdx.x; i < WORDS_REGION / 4; i += (int)blockDim.x) {
    uint32_t off = (uint32_t)i * 16u;
    uint32_t ldsAddr = ldsBase + off;
    asm volatile("global_load_async_to_lds_b128 %0, %1, %2 offset:0"
                 :: "v"(ldsAddr), "v"(off), "s"(src)
                 : "memory");
  }
}
__device__ __forceinline__ void wait_async_all() {
  asm volatile("s_wait_asynccnt 0" ::: "memory");
}

// ---------------------------------------------------------------------------
// RNG for rank-pair sampling (rejection sampling uniform over core1 pixels)
// ---------------------------------------------------------------------------
__device__ __forceinline__ unsigned hashu(unsigned x) {
  x ^= x >> 16; x *= 0x7feb352du;
  x ^= x >> 15; x *= 0x846ca68bu;
  x ^= x >> 16;
  return x;
}

__device__ __forceinline__ unsigned sample_idx(const unsigned* bm, unsigned g,
                                               unsigned pair, unsigned salt) {
  unsigned idx = 0;
  for (unsigned a = 0; a < 64; ++a) {
    unsigned h = hashu(g * 0x9E3779B9u + pair * 0x85EBCA6Bu + a * 0xC2B2AE35u + salt);
    idx = h & (HW_ - 1u);
    if ((bm[idx >> 5] >> (idx & 31u)) & 1u) return idx;
  }
  return idx;  // fallback (region empty after erosion): ~uniform over HW
}

// ---------------------------------------------------------------------------
// Kernels
// ---------------------------------------------------------------------------
__global__ void k_init(int* counts, float* stats, float* partials) {
  int t = (int)threadIdx.x;
  if (t < 3 * NREG)   counts[t]   = 0;
  if (t < NREG * 12)  stats[t]    = 0.0f;
  if (t < 256)        partials[t] = 0.0f;
}

// mean(|pred - target|) partials: 256 blocks x 256 threads, b128 loads.
__global__ void k_mean(const float4* __restrict__ pred4,
                       const float4* __restrict__ target4,
                       float* __restrict__ partials) {
  const int n4 = (B_ * HW_) / 4;   // 1048576 float4 elements
  float s = 0.0f;
  for (int i = (int)(blockIdx.x * blockDim.x + threadIdx.x); i < n4;
       i += (int)(gridDim.x * blockDim.x)) {
    float4 p = pred4[i];
    float4 t = target4[i];
    s += fabsf(p.x - t.x) + fabsf(p.y - t.y) + fabsf(p.z - t.z) + fabsf(p.w - t.w);
  }
  __shared__ float red[8];
  float r = block_reduce_sum(s, red);
  if (threadIdx.x == 0) partials[blockIdx.x] = r;
}

// Build per-region bitmasks from lab (labels 5, 8, 13) + region popcounts.
// Each thread packs one 32-column word via eight int4 (b128) loads.
__global__ void k_bits(const int4* __restrict__ lab4,
                       unsigned* __restrict__ regionBits,
                       int* __restrict__ nReg) {
  int idx = (int)(blockIdx.x * blockDim.x + threadIdx.x);
  if (idx >= B_ * H_ * WPR) return;
  int w = idx % WPR;
  int y = (idx / WPR) % H_;
  int b = idx / (WPR * H_);
  const int4* base = lab4 + ((size_t)b * HW_ + (size_t)y * W_ + (size_t)w * 32) / 4;
  unsigned m0 = 0, m1 = 0, m2 = 0;
  #pragma unroll
  for (int q = 0; q < 8; ++q) {
    int4 v = base[q];
    int j = q * 4;
    m0 |= ((unsigned)(v.x == 5)  << (j + 0)) | ((unsigned)(v.y == 5)  << (j + 1)) |
          ((unsigned)(v.z == 5)  << (j + 2)) | ((unsigned)(v.w == 5)  << (j + 3));
    m1 |= ((unsigned)(v.x == 8)  << (j + 0)) | ((unsigned)(v.y == 8)  << (j + 1)) |
          ((unsigned)(v.z == 8)  << (j + 2)) | ((unsigned)(v.w == 8)  << (j + 3));
    m2 |= ((unsigned)(v.x == 13) << (j + 0)) | ((unsigned)(v.y == 13) << (j + 1)) |
          ((unsigned)(v.z == 13) << (j + 2)) | ((unsigned)(v.w == 13) << (j + 3));
  }
  int rowOff = y * WPR + w;
  regionBits[(size_t)(b * 3 + 0) * WORDS_REGION + rowOff] = m0;
  regionBits[(size_t)(b * 3 + 1) * WORDS_REGION + rowOff] = m1;
  regionBits[(size_t)(b * 3 + 2) * WORDS_REGION + rowOff] = m2;
  atomicAdd(&nReg[b * 3 + 0], __popc(m0));
  atomicAdd(&nReg[b * 3 + 1], __popc(m1));
  atomicAdd(&nReg[b * 3 + 2], __popc(m2));
}

// 3x3 binary erosion on bitmaps; out-of-bounds treated as 1 (inf-pad min).
__global__ void k_erode(const unsigned* __restrict__ in,
                        unsigned* __restrict__ out,
                        int* __restrict__ cnt) {
  int idx = (int)(blockIdx.x * blockDim.x + threadIdx.x);
  if (idx >= NREG * WORDS_REGION) return;
  int g   = idx / WORDS_REGION;
  int rem = idx % WORDS_REGION;
  int y = rem >> 4, w = rem & 15;
  const unsigned* base = in + (size_t)g * WORDS_REGION;
  unsigned v = 0xFFFFFFFFu;
  #pragma unroll
  for (int dy = -1; dy <= 1; ++dy) {
    int yy = y + dy;
    if (yy < 0 || yy >= H_) continue;   // OOB row -> all ones -> no-op
    const unsigned* row = base + yy * WPR;
    unsigned c     = row[w];
    unsigned left  = (c << 1) | ((w > 0)  ? (row[w - 1] >> 31) : 1u);
    unsigned right = (c >> 1) | ((w < 15) ? (row[w + 1] << 31) : 0x80000000u);
    v &= c & left & right;
  }
  out[idx] = v;
  atomicAdd(&cnt[g], __popc(v));
}

// Per-region statistics: one 256-thread block per region.  core1/core2 bitmaps
// are staged into LDS with async tensor-path loads.
__global__ void k_stats(const float* __restrict__ pred,
                        const unsigned* __restrict__ regionBits,
                        const unsigned* __restrict__ e1Bits,
                        const unsigned* __restrict__ e2Bits,
                        const int* __restrict__ s1cnt,
                        const int* __restrict__ s2cnt,
                        float* __restrict__ stats) {
  __shared__ unsigned s_c1[WORDS_REGION];   // 32 KB
  __shared__ unsigned s_c2[WORDS_REGION];   // 32 KB

  const int g = (int)blockIdx.x;
  const int b = g / 3;
  const unsigned* c1src = ((s1cnt[g] < 2) ? regionBits : e1Bits) + (size_t)g * WORDS_REGION;
  const unsigned* c2src = (s2cnt[g] < 3) ? c1src
                                         : (e2Bits + (size_t)g * WORDS_REGION);

  async_copy_bitmap(s_c1, c1src);
  async_copy_bitmap(s_c2, c2src);
  wait_async_all();
  __syncthreads();

  const float* predB = pred + (size_t)b * HW_;
  float n1 = 0.f, n2 = 0.f, sp = 0.f, sp2 = 0.f;
  float sdx = 0.f, cdx = 0.f, sdy = 0.f, cdy = 0.f;

  for (int i = (int)threadIdx.x; i < WORDS_REGION; i += (int)blockDim.x) {
    int y = i >> 4, w = i & 15;
    unsigned c1 = s_c1[i];
    unsigned c2 = s_c2[i];
    n1 += (float)__popc(c1);
    n2 += (float)__popc(c2);
    int pixBase = y * W_ + w * 32;

    unsigned m = c2;                      // mean / variance accumulation
    while (m) {
      int j = __ffs((int)m) - 1; m &= m - 1;
      float p = predB[pixBase + j];
      sp += p; sp2 += p * p;
    }
    unsigned nextLow = (w < 15) ? (s_c2[i + 1] & 1u) : 0u;   // horizontal pairs
    unsigned mx = c2 & ((c2 >> 1) | (nextLow << 31));
    cdx += (float)__popc(mx);
    m = mx;
    while (m) {
      int j = __ffs((int)m) - 1; m &= m - 1;
      sdx += fabsf(predB[pixBase + j + 1] - predB[pixBase + j]);
    }
    unsigned below = (y < H_ - 1) ? s_c2[i + WPR] : 0u;      // vertical pairs
    unsigned my = c2 & below;
    cdy += (float)__popc(my);
    m = my;
    while (m) {
      int j = __ffs((int)m) - 1; m &= m - 1;
      sdy += fabsf(predB[pixBase + W_ + j] - predB[pixBase + j]);
    }
  }
  __syncthreads();

  float* red = (float*)s_c1;   // bitmaps no longer needed: reuse as scratch
  float vals[8] = {n1, n2, sp, sp2, sdx, cdx, sdy, cdy};
  #pragma unroll
  for (int k = 0; k < 8; ++k) {
    float r = block_reduce_sum(vals[k], red);
    if (threadIdx.x == 0) stats[g * 12 + k] = r;
  }
}

// Rank hinge term: one 128-thread block per region, 100 sampled pairs.
__global__ void k_rank(const float* __restrict__ pred,
                       const float* __restrict__ img1,
                       const float* __restrict__ img2,
                       const unsigned* __restrict__ regionBits,
                       const unsigned* __restrict__ e1Bits,
                       const int* __restrict__ s1cnt,
                       float* __restrict__ stats) {
  const int g = (int)blockIdx.x;
  const int b = g / 3;
  const unsigned* bm = ((s1cnt[g] < 2) ? regionBits : e1Bits) + (size_t)g * WORDS_REGION;
  const float* predB = pred + (size_t)b * HW_;
  const float* t1    = img1 + (size_t)b * HW_;
  const float* t2    = img2 + (size_t)b * HW_;

  float hs = 0.0f, pv = 0.0f;
  int pair = (int)threadIdx.x;
  if (pair < 100) {
    unsigned ii = sample_idx(bm, (unsigned)g, (unsigned)pair, 0x00012345u);
    unsigned jj = sample_idx(bm, (unsigned)g, (unsigned)pair, 0x000ABCDEu);
    float xi = 0.5f * (1.0f - t1[ii]) + 0.5f * t2[ii];
    float xj = 0.5f * (1.0f - t1[jj]) + 0.5f * t2[jj];
    float s  = (xi > xj) ? 1.0f : ((xi < xj) ? -1.0f : 0.0f);
    float ok = ((ii != jj) && (s != 0.0f)) ? 1.0f : 0.0f;
    float h  = fmaxf(0.0f, -(predB[ii] - predB[jj]) * s);
    hs = h * ok;
    pv = ok;
  }
  __shared__ float red[4];
  float r0 = block_reduce_sum(hs, red);
  float r1 = block_reduce_sum(pv, red);
  if (threadIdx.x == 0) {
    stats[g * 12 + 8] = r0;
    stats[g * 12 + 9] = r1;
  }
}

// Final scalar combine (tiny, single thread).
__global__ void k_final(const float* __restrict__ partials,
                        const int* __restrict__ counts,
                        const float* __restrict__ stats,
                        float* __restrict__ out) {
  if (threadIdx.x != 0 || blockIdx.x != 0) return;
  const float tgtcv[3] = {0.077f, 0.227f, 0.348f};

  float meanSum = 0.0f;
  for (int i = 0; i < 256; ++i) meanSum += partials[i];
  float loss_mean = meanSum / (float)(B_ * HW_);

  float sv = 0.f, svc = 0.f, rv = 0.f, rvc = 0.f, mv = 0.f, mvc = 0.f;
  for (int g = 0; g < NREG; ++g) {
    int   nreg = counts[g];               // nReg[g]
    float n1   = stats[g * 12 + 0];
    float n2   = stats[g * 12 + 1];
    float sp   = stats[g * 12 + 2];
    float sp2  = stats[g * 12 + 3];
    float sdx  = stats[g * 12 + 4];
    float cdx  = stats[g * 12 + 5];
    float sdy  = stats[g * 12 + 6];
    float cdy  = stats[g * 12 + 7];
    float hng  = stats[g * 12 + 8];
    float npv  = stats[g * 12 + 9];

    float n2s = fmaxf(n2, 1.0f);
    float mu  = sp / n2s;
    float var = fmaxf(sp2 / n2s - mu * mu, 0.0f);
    float cv  = sqrtf(var) / (fabsf(mu) + 1e-6f);
    float stdv = fabsf(cv - tgtcv[g % 3]);
    if (nreg >= 3 && n2 >= 3.0f) { sv += stdv; svc += 1.0f; }

    float rankv = hng / fmaxf(npv, 1.0f);
    if (nreg >= 2 && n1 >= 2.0f && npv > 0.0f) { rv += rankv; rvc += 1.0f; }

    float mean_x = sdx / fmaxf(cdx, 1.0f);
    float mean_y = sdy / fmaxf(cdy, 1.0f);
    float hx = (cdx > 0.0f) ? 1.0f : 0.0f;
    float hy = (cdy > 0.0f) ? 1.0f : 0.0f;
    float npres = hx + hy;
    float smoothv = (mean_x * hx + mean_y * hy) / fmaxf(npres, 1.0f);
    if (nreg >= 3 && n2 >= 3.0f && npres > 0.0f) { mv += smoothv; mvc += 1.0f; }
  }
  float loss_std    = (svc > 0.0f) ? sv / svc : 0.0f;
  float loss_rank   = (rvc > 0.0f) ? rv / rvc : 0.0f;
  float loss_smooth = (mvc > 0.0f) ? mv / mvc : 0.0f;
  float loss_stat   = 0.5f * loss_mean + 0.5f * loss_std;
  out[0] = loss_stat + loss_rank + loss_smooth;
}

// ---------------------------------------------------------------------------
// Host-side launch
// ---------------------------------------------------------------------------
extern "C" void kernel_launch(void* const* d_in, const int* in_sizes, int n_in,
                              void* d_out, int out_size, void* d_ws, size_t ws_size,
                              hipStream_t stream) {
  (void)in_sizes; (void)n_in; (void)out_size; (void)ws_size;
  const float* pred   = (const float*)d_in[0];
  const float* target = (const float*)d_in[1];
  const int*   lab    = (const int*)  d_in[2];
  const float* img1   = (const float*)d_in[3];
  const float* img2   = (const float*)d_in[4];
  float* out = (float*)d_out;

  // Workspace layout
  char* ws = (char*)d_ws;
  unsigned* regionBits = (unsigned*)(ws);
  unsigned* e1Bits     = (unsigned*)(ws + (size_t)NREG * WORDS_REGION * 4);
  unsigned* e2Bits     = (unsigned*)(ws + (size_t)2 * NREG * WORDS_REGION * 4);
  int*      counts     = (int*)     (ws + (size_t)3 * NREG * WORDS_REGION * 4);
  int*      s1         = counts + NREG;
  int*      s2         = counts + 2 * NREG;
  float*    stats      = (float*)(counts + 3 * NREG);
  float*    partials   = stats + NREG * 12;

  k_init<<<1, 1024, 0, stream>>>(counts, stats, partials);
  k_bits<<<(B_ * H_ * WPR) / 256, 256, 0, stream>>>((const int4*)lab, regionBits, counts);
  k_mean<<<256, 256, 0, stream>>>((const float4*)pred, (const float4*)target, partials);
  k_erode<<<(NREG * WORDS_REGION) / 256, 256, 0, stream>>>(regionBits, e1Bits, s1);
  k_erode<<<(NREG * WORDS_REGION) / 256, 256, 0, stream>>>(e1Bits, e2Bits, s2);
  k_stats<<<NREG, 256, 0, stream>>>(pred, regionBits, e1Bits, e2Bits, s1, s2, stats);
  k_rank<<<NREG, 128, 0, stream>>>(pred, img1, img2, regionBits, e1Bits, s1, stats);
  k_final<<<1, 32, 0, stream>>>(partials, counts, stats, out);
}
